// JointVectorQuantizerEMA_LowMem_61649960567435
// MI455X (gfx1250) — matrix operand order repaired and checked by reading, not compiled, not run on hardware
//
#include <hip/hip_runtime.h>
#include <hip/hip_bf16.h>

typedef __attribute__((ext_vector_type(16))) __bf16 bf16x16;
typedef __attribute__((ext_vector_type(8)))  __bf16 bf16x8;
typedef __attribute__((ext_vector_type(8)))  float  f32x8;

constexpr int Bb = 32, Dd = 256, Hh = 32, Ww = 32;
constexpr int HW = Hh * Ww;            // 1024
constexpr int Nn = Bb * HW;            // 32768 vectors
constexpr int Kk = 8192;               // codebook size
constexpr float DECAY = 0.99f;
constexpr float EPS = 1e-5f;
constexpr float CCOST = 0.25f;

// ---------------- workspace layout (bytes) ----------------
constexpr size_t WS_XB    = 0;          // flat_x bf16  N*D*2 = 16,777,216
constexpr size_t WS_ET    = 16777216;   // emb_T  bf16  D*K*2 =  4,194,304
constexpr size_t WS_E2    = 20971520;   // e2     f32   K*4
constexpr size_t WS_CODES = 21004288;   // codes  i32   N*4
constexpr size_t WS_CNT   = 21135360;   // counts f32   K*4        } zeroed
constexpr size_t WS_ESUM  = 21168128;   // esum   f32   K*D*4      } every
constexpr size_t WS_SCAL  = 29556736;   // scal   f32   2*4        } launch
constexpr int ZERO_FLOATS = Kk + Kk * Dd + 2;   // counts + esum + scalars (contiguous)

// ---------------- output layout (floats) ----------------
constexpr int O_XQ   = 0;                       // B*D*H*W
constexpr int O_LOSS = 8388608;                 // 1
constexpr int O_COD  = 8388609;                 // N
constexpr int O_NCS  = 8421377;                 // K
constexpr int O_NEA  = 8429569;                 // K*D
constexpr int O_EN   = 10526721;                // K*D

__global__ void k_zero(float* p, int n) {
  int i = blockIdx.x * blockDim.x + threadIdx.x;
  if (i < n) p[i] = 0.0f;
}

// x [B,D,H,W] f32  ->  flat_x [N,D] bf16  (N index = b*HW + hw), LDS tile transpose
__global__ void k_prep_x(const float* __restrict__ x, __hip_bfloat16* __restrict__ xb) {
  __shared__ float tile[32][33];
  int bid = blockIdx.x;
  int n0 = (bid % (Nn / 32)) * 32;          // 32 consecutive n (same b: HW%32==0)
  int d0 = (bid / (Nn / 32)) * 32;
  int b  = n0 / HW;
  int hw0 = n0 % HW;
  int tx = threadIdx.x & 31;
  int ty = threadIdx.x >> 5;                // 0..7
#pragma unroll
  for (int j = 0; j < 32; j += 8)           // coalesced read along hw
    tile[ty + j][tx] = x[(b * Dd + d0 + ty + j) * HW + hw0 + tx];
  __syncthreads();
#pragma unroll
  for (int j = 0; j < 32; j += 8)           // coalesced write along d
    xb[(size_t)(n0 + ty + j) * Dd + d0 + tx] = __float2bfloat16(tile[tx][ty + j]);
}

// emb [K,D] f32 -> emb_T [D,K] bf16, LDS tile transpose
__global__ void k_prep_embT(const float* __restrict__ emb, __hip_bfloat16* __restrict__ et) {
  __shared__ float tile[32][33];
  int bid = blockIdx.x;
  int k0 = (bid % (Kk / 32)) * 32;
  int d0 = (bid / (Kk / 32)) * 32;
  int tx = threadIdx.x & 31;
  int ty = threadIdx.x >> 5;
#pragma unroll
  for (int j = 0; j < 32; j += 8)           // coalesced read along d
    tile[ty + j][tx] = emb[(size_t)(k0 + ty + j) * Dd + d0 + tx];
  __syncthreads();
#pragma unroll
  for (int j = 0; j < 32; j += 8)           // coalesced write along k
    et[(size_t)(d0 + ty + j) * Kk + k0 + tx] = __float2bfloat16(tile[tx][ty + j]);
}

// e2[k] = sum_d emb[k,d]^2 (exact f32)
__global__ void k_e2(const float* __restrict__ emb, float* __restrict__ e2) {
  int k = blockIdx.x * blockDim.x + threadIdx.x;
  const float4* p = (const float4*)(emb + (size_t)k * Dd);
  float s = 0.0f;
#pragma unroll 4
  for (int i = 0; i < Dd / 4; ++i) {
    float4 v = p[i];
    s = fmaf(v.x, v.x, s); s = fmaf(v.y, v.y, s);
    s = fmaf(v.z, v.z, s); s = fmaf(v.w, v.w, s);
  }
  e2[k] = s;
}

// ---- main WMMA argmin -------------------------------------------------------
// Block = 4 waves, each owning 32 rows (2 x 16-row tiles). All waves scan K in
// lockstep. Each 16-code chunk of emb_T (256 dims x 16 codes bf16 = 8 KB) is
// staged through a TRIPLE-buffered LDS tile with global_load_async_to_lds_b128
// (ASYNCcnt DMA, in-order completion): chunk i+2 is issued while chunk i is
// computed, so DMA runs a full chunk ahead of the 16 WMMAs. B fragments are
// preloaded into a register array (pipelined ds_loads, no per-step dscnt-0
// stalls) and e2 is software-pipelined one chunk ahead in a register.
__global__ __launch_bounds__(128) void k_argmin(
    const __bf16* __restrict__ xb, const __bf16* __restrict__ et,
    const float* __restrict__ e2, int* __restrict__ codes,
    float* __restrict__ out_codes)
{
  constexpr int CHUNK_B = Dd * 16 * (int)sizeof(__bf16);   // 8 KB
  __shared__ __bf16 sbuf[3][Dd * 16];                      // 3-deep chunk pipeline

  const int tid  = threadIdx.x;
  const int lane = tid & 31;
  const int wave = tid >> 5;
  const int rowBase = (blockIdx.x * 4 + wave) * 32;
  const int half = lane >> 4;               // C-fragment half: rows 0-7 vs 8-15
  const int eoff = half * 8;                // A-layout K-offset for this lane half
  const int colSel = lane & 15;             // this lane's code column in a chunk

  // A fragments: 2 tiles x 8 K-slices of 32, resident for the whole K scan
  bf16x16 a[2][8];
#pragma unroll
  for (int t = 0; t < 2; ++t) {
    const __bf16* rp = xb + (size_t)(rowBase + t * 16 + colSel) * Dd;
#pragma unroll
    for (int f = 0; f < 8; ++f) {
      const __bf16* base = rp + f * 32;
      bf16x8 lo = *(const bf16x8*)(base + eoff);        // K = eoff..eoff+7
      bf16x8 hi = *(const bf16x8*)(base + 16 + eoff);   // K = 16+eoff..
      bf16x16 av;
#pragma unroll
      for (int j = 0; j < 8; ++j) { av[j] = lo[j]; av[j + 8] = hi[j]; }
      a[t][f] = av;
    }
  }

  // Per-thread async-copy addresses: idx = tid + r*128 in [0,512):
  // d = idx>>1 (0..255), h = idx&1 selects the 16-byte half of the 32 B row.
  unsigned laddr0[4];
  unsigned long long goff0[4];
#pragma unroll
  for (int r = 0; r < 4; ++r) {
    int idx = tid + r * 128;
    int d = idx >> 1, h = idx & 1;
    laddr0[r] = (unsigned)(size_t)(const void*)&sbuf[0][d * 16 + h * 8];
    goff0[r]  = (unsigned long long)(size_t)(const void*)(et + (size_t)d * Kk + h * 8);
  }

  // prologue: DMA chunks 0 and 1 (4 wave-instructions each = 8 KB via 4 waves)
#pragma unroll
  for (int p = 0; p < 2; ++p) {
#pragma unroll
    for (int r = 0; r < 4; ++r) {
      unsigned la = laddr0[r] + (unsigned)(p * CHUNK_B);
      unsigned long long ga = goff0[r] + (unsigned long long)(p * 16) * sizeof(__bf16);
      asm volatile("global_load_async_to_lds_b128 %0, %1, off"
                   :: "v"(la), "v"(ga) : "memory");
    }
  }

  float best0[8], best1[8];
  int   idx0[8],  idx1[8];
#pragma unroll
  for (int r = 0; r < 8; ++r) {
    best0[r] = 3.4e38f; best1[r] = 3.4e38f; idx0[r] = 0; idx1[r] = 0;
  }

  float e2c = e2[colSel];                   // e2 for chunk 0, pipelined ahead

  for (int c0 = 0; c0 < Kk; c0 += 16) {
    const int chunk = c0 >> 4;
    const int cur = chunk - (chunk / 3) * 3;          // chunk % 3

    // issue e2 for the NEXT chunk now; consumed next iteration (latency hidden)
    int nIdx = c0 + 16 + colSel;
    float e2n = e2[nIdx < Kk ? nIdx : (Kk - 1)];

    // chunk i's 4 copies done (in-order ASYNCcnt); chunk i+1's may stay in flight
    if (c0 + 16 < Kk) asm volatile("s_wait_asynccnt 0x4" ::: "memory");
    else              asm volatile("s_wait_asynccnt 0x0" ::: "memory");
    __syncthreads();                                  // everyone's chunk i landed

    // kick off DMA of chunk i+2 into the buffer consumed in the previous iteration
    if (c0 + 32 < Kk) {
      const int nxt = (chunk + 2) - ((chunk + 2) / 3) * 3;
#pragma unroll
      for (int r = 0; r < 4; ++r) {
        unsigned la = laddr0[r] + (unsigned)(nxt * CHUNK_B);
        unsigned long long ga = goff0[r] + (unsigned long long)(c0 + 32) * sizeof(__bf16);
        asm volatile("global_load_async_to_lds_b128 %0, %1, off"
                     :: "v"(la), "v"(ga) : "memory");
      }
    }

    // preload all 8 B fragments -> pipelined ds_loads, no per-step full drains
    const __bf16* sb = &sbuf[0][0] + (size_t)cur * (Dd * 16);
    bf16x16 bf[8];
#pragma unroll
    for (int f = 0; f < 8; ++f) {
#pragma unroll
      for (int j = 0; j < 8; ++j) {
        bf[f][j]     = sb[(f * 32 + eoff + j) * 16 + colSel];
        bf[f][j + 8] = sb[(f * 32 + 16 + eoff + j) * 16 + colSel];
      }
    }

    f32x8 acc0 = {0.f, 0.f, 0.f, 0.f, 0.f, 0.f, 0.f, 0.f};
    f32x8 acc1 = {0.f, 0.f, 0.f, 0.f, 0.f, 0.f, 0.f, 0.f};
#pragma unroll
    for (int f = 0; f < 8; ++f) {
      acc0 = __builtin_amdgcn_wmma_f32_16x16x32_bf16(false, a[0][f], false, bf[f],
                                                     (short)0, acc0, false, false);
      acc1 = __builtin_amdgcn_wmma_f32_16x16x32_bf16(false, a[1][f], false, bf[f],
                                                     (short)0, acc1, false, false);
    }

    const int cidx = c0 + colSel;
#pragma unroll
    for (int r = 0; r < 8; ++r) {             // branch-free: keeps EXEC all-1s
      float v0 = fmaf(-2.0f, acc0[r], e2c);
      float v1 = fmaf(-2.0f, acc1[r], e2c);
      bool b0 = v0 < best0[r];
      bool b1 = v1 < best1[r];
      best0[r] = b0 ? v0 : best0[r];  idx0[r] = b0 ? cidx : idx0[r];
      best1[r] = b1 ? v1 : best1[r];  idx1[r] = b1 ? cidx : idx1[r];
    }
    e2c = e2n;
  }

  // cross-lane argmin within each 16-lane group; tie -> lower index (first min)
#pragma unroll
  for (int r = 0; r < 8; ++r) {
#pragma unroll
    for (int off = 8; off >= 1; off >>= 1) {
      float ov0 = __shfl_xor(best0[r], off, 32); int oi0 = __shfl_xor(idx0[r], off, 32);
      float ov1 = __shfl_xor(best1[r], off, 32); int oi1 = __shfl_xor(idx1[r], off, 32);
      bool t0 = (ov0 < best0[r]) || (ov0 == best0[r] && oi0 < idx0[r]);
      bool t1 = (ov1 < best1[r]) || (ov1 == best1[r] && oi1 < idx1[r]);
      best0[r] = t0 ? ov0 : best0[r];  idx0[r] = t0 ? oi0 : idx0[r];
      best1[r] = t1 ? ov1 : best1[r];  idx1[r] = t1 ? oi1 : idx1[r];
    }
  }
  if ((lane & 15) == 0) {
#pragma unroll
    for (int r = 0; r < 8; ++r) {
      int row0 = rowBase + half * 8 + r;        // tile 0: rows rowBase..+15
      int row1 = rowBase + 16 + half * 8 + r;   // tile 1
      codes[row0] = idx0[r];  out_codes[row0] = (float)idx0[r];
      codes[row1] = idx1[r];  out_codes[row1] = (float)idx1[r];
    }
  }
}

// gather x_q, write x_q_st in [B,D,H,W] layout, scatter-add stats, accumulate loss
__global__ void k_scatter(const float* __restrict__ x, const float* __restrict__ emb,
                          const int* __restrict__ codes, float* __restrict__ counts,
                          float* __restrict__ esum, float* __restrict__ out_xq,
                          float* __restrict__ loss_acc) {
  int n = blockIdx.x * blockDim.x + threadIdx.x;
  int c  = codes[n];
  int b  = n / HW;
  int hw = n % HW;
  atomicAdd(&counts[c], 1.0f);
  float lsum = 0.0f;
  for (int d = 0; d < Dd; ++d) {
    float xv = x[(size_t)(b * Dd + d) * HW + hw];     // coalesced across threads
    float qv = emb[(size_t)c * Dd + d];
    atomicAdd(&esum[(size_t)c * Dd + d], xv);
    float diff = qv - xv;
    out_xq[(size_t)(b * Dd + d) * HW + hw] = xv + diff;  // x + sg(x_q - x), as ref
    lsum = fmaf(diff, diff, lsum);
  }
#pragma unroll
  for (int off = 16; off; off >>= 1) lsum += __shfl_xor(lsum, off, 32);
  if ((threadIdx.x & 31) == 0) atomicAdd(loss_acc, lsum);
}

__global__ void k_cluster(const float* __restrict__ cs, const float* __restrict__ counts,
                          float* __restrict__ out_ncs, float* __restrict__ n_acc) {
  int k = blockIdx.x * blockDim.x + threadIdx.x;
  float ncs = cs[k] * DECAY + (1.0f - DECAY) * counts[k];
  out_ncs[k] = ncs;
  float s = ncs;
#pragma unroll
  for (int off = 16; off; off >>= 1) s += __shfl_xor(s, off, 32);
  if ((threadIdx.x & 31) == 0) atomicAdd(n_acc, s);
}

__global__ void k_avg(const float* __restrict__ ea, const float* __restrict__ esum,
                      float* __restrict__ out_nea) {
  int i = blockIdx.x * blockDim.x + threadIdx.x;
  out_nea[i] = ea[i] * DECAY + (1.0f - DECAY) * esum[i];
}

__global__ void k_final(const float* __restrict__ out_ncs, const float* __restrict__ out_nea,
                        const float* __restrict__ scal, float* __restrict__ out_en,
                        float* __restrict__ out_loss) {
  int i = blockIdx.x * blockDim.x + threadIdx.x;
  int k = i / Dd;
  float ntot = scal[1];
  float smoothed = (out_ncs[k] + EPS) / (ntot + (float)Kk * EPS) * ntot;
  out_en[i] = out_nea[i] / smoothed;
  if (i == 0) out_loss[0] = (1.0f + CCOST) * scal[0] / (float)((size_t)Nn * Dd);
}

extern "C" void kernel_launch(void* const* d_in, const int* in_sizes, int n_in,
                              void* d_out, int out_size, void* d_ws, size_t ws_size,
                              hipStream_t stream) {
  const float* x   = (const float*)d_in[0];
  const float* emb = (const float*)d_in[1];
  const float* cs  = (const float*)d_in[2];
  const float* ea  = (const float*)d_in[3];

  char* ws = (char*)d_ws;
  __hip_bfloat16* xb = (__hip_bfloat16*)(ws + WS_XB);
  __hip_bfloat16* et = (__hip_bfloat16*)(ws + WS_ET);
  float* e2     = (float*)(ws + WS_E2);
  int*   codes  = (int*)  (ws + WS_CODES);
  float* counts = (float*)(ws + WS_CNT);
  float* esum   = (float*)(ws + WS_ESUM);
  float* scal   = (float*)(ws + WS_SCAL);   // [0]=loss accum, [1]=n accum

  float* out = (float*)d_out;

  // 1) zero accumulators (contiguous: counts, esum, scalars)
  k_zero<<<(ZERO_FLOATS + 255) / 256, 256, 0, stream>>>(counts, ZERO_FLOATS);
  // 2) data prep
  k_prep_x   <<<(Nn / 32) * (Dd / 32), 256, 0, stream>>>(x, xb);
  k_prep_embT<<<(Kk / 32) * (Dd / 32), 256, 0, stream>>>(emb, et);
  k_e2       <<<Kk / 256, 256, 0, stream>>>(emb, e2);
  // 3) WMMA nearest-code search (async-LDS triple-buffered B)
  k_argmin   <<<Nn / 128, 128, 0, stream>>>((const __bf16*)xb, (const __bf16*)et,
                                            e2, codes, out + O_COD);
  // 4) gather/scatter epilogue
  k_scatter  <<<Nn / 256, 256, 0, stream>>>(x, emb, codes, counts, esum,
                                            out + O_XQ, scal + 0);
  // 5) EMA updates
  k_cluster  <<<Kk / 256, 256, 0, stream>>>(cs, counts, out + O_NCS, scal + 1);
  k_avg      <<<(Kk * Dd) / 256, 256, 0, stream>>>(ea, esum, out + O_NEA);
  // 6) normalize + loss
  k_final    <<<(Kk * Dd) / 256, 256, 0, stream>>>(out + O_NCS, out + O_NEA, scal,
                                                   out + O_EN, out + O_LOSS);
}
